// SelfAttention_77884936945946
// MI455X (gfx1250) — matrix operand order repaired and checked by reading, not compiled, not run on hardware
//
#include <hip/hip_runtime.h>
#include <hip/hip_bf16.h>
#include <stdint.h>

// Problem constants (from reference)
#define E_DIM  1024
#define NHEADS 16
#define HDIM   64
#define BATCH  4
#define SEQ    2048
#define MROWS  (BATCH * SEQ)   // 8192

typedef __bf16 bf16_t;
typedef __attribute__((ext_vector_type(16))) __bf16 v16bf;
typedef __attribute__((ext_vector_type(8)))  __bf16 v8bf;
typedef __attribute__((ext_vector_type(8)))  float  v8f;

// ---------------------------------------------------------------------------
// helpers
// ---------------------------------------------------------------------------
static __device__ __forceinline__ v8f wmma_bf16(v16bf a, v16bf b, v8f c) {
  // D = A(16x32) * B(32x16) + C, f32 accumulate
  return __builtin_amdgcn_wmma_f32_16x16x32_bf16(false, a, false, b,
                                                 (short)0, c, false, false);
}

// hardware f32->bf16 conversion (compiler lowers to v_cvt_*bf16* on gfx1250)
static __device__ __forceinline__ bf16_t f2bf(float f) { return (bf16_t)f; }

// ---- CDNA5 async global->LDS copy (ASYNCcnt path) -------------------------
#if __has_builtin(__builtin_amdgcn_global_load_async_to_lds_b128)
#define HAVE_ASYNC_LDS 1
#endif

// The builtin's pointer params are GCC-vector int4 pointers:
//   arg0: AS1 (global) int4*   arg1: AS3 (LDS) int4*   arg2/3: imm offset/cpol
typedef int v4i_vs __attribute__((vector_size(16)));
typedef __attribute__((address_space(1))) v4i_vs* gptr_v4i;
typedef __attribute__((address_space(3))) v4i_vs* lptr_v4i;

static __device__ __forceinline__ void async_copy16(void* lds, const void* g) {
#ifdef HAVE_ASYNC_LDS
  __builtin_amdgcn_global_load_async_to_lds_b128((gptr_v4i)g, (lptr_v4i)lds,
                                                 0, 0);
#else
  *(uint4*)lds = *(const uint4*)g;
#endif
}

static __device__ __forceinline__ void async_wait() {
#ifdef HAVE_ASYNC_LDS
#if __has_builtin(__builtin_amdgcn_s_wait_asynccnt)
  __builtin_amdgcn_s_wait_asynccnt(0);
#else
  asm volatile("s_wait_asynccnt 0x0" ::: "memory");
#endif
#endif
}

// A fragment (16x32 MxK), source row-major [row][k], `ld` elements per row.
// Lane L holds row M=L%16; elements 0..7 -> K = 8*(L/16)+0..7,
// elements 8..15 -> K = 16 + 8*(L/16)+0..7 (ISA 16-bit A layout).
static __device__ __forceinline__ v16bf frag_a(const bf16_t* p, int ld, int lane) {
  const bf16_t* q = p + (lane & 15) * ld + ((lane >> 4) << 3);
  union { v16bf v; v8bf h[2]; } u;
  u.h[0] = *(const v8bf*)(q);
  u.h[1] = *(const v8bf*)(q + 16);
  return u.v;
}

// B fragment (32x16 KxN), source stored "transposed": Bsrc[n][k] row-major.
// Lane L holds column N=L%16; elements 0..7 -> K = 8*(L/16)+0..7,
// elements 8..15 -> K = 16 + 8*(L/16)+0..7 (same K order as frag_a).
static __device__ __forceinline__ v16bf frag_b(const bf16_t* p, int ld, int lane) {
  const bf16_t* q = p + (lane & 15) * ld + ((lane >> 4) << 3);
  union { v16bf v; v8bf h[2]; } u;
  u.h[0] = *(const v8bf*)(q);
  u.h[1] = *(const v8bf*)(q + 16);
  return u.v;
}

// ---------------------------------------------------------------------------
// Kernel 0: transpose conv weight  Wc[E][E][3] f32 -> WcT[3][E][E] bf16
// ---------------------------------------------------------------------------
__global__ __launch_bounds__(256) void prep_wc(const float* __restrict__ Wc,
                                               bf16_t* __restrict__ WcT) {
  int idx = blockIdx.x * 256 + threadIdx.x;          // over E*E
  int eo = idx / E_DIM, ei = idx % E_DIM;
  #pragma unroll
  for (int k = 0; k < 3; ++k)
    WcT[(size_t)k * E_DIM * E_DIM + (size_t)eo * E_DIM + ei] =
        f2bf(Wc[((size_t)eo * E_DIM + ei) * 3 + k]);
}

// ---------------------------------------------------------------------------
// Kernel 1: QKV projection  y = x @ W.T  (M=8192, N=1024, K=1024) for 3 W's.
// Block tile 128x128, K-chunk 32, 8 waves in 4x2, each wave 32x64 (2x4 WMMA).
// Scatters Q,K -> [B,H,S,D] bf16 ; V -> transposed [B,H,D,S] bf16.
// (staging stays in VGPRs here because it must convert f32->bf16 in flight)
// ---------------------------------------------------------------------------
#define BM 128
#define BN 128
#define BK 32

__global__ __launch_bounds__(256) void qkv_gemm(
    const float* __restrict__ x,
    const float* __restrict__ Wq, const float* __restrict__ Wk,
    const float* __restrict__ Wv,
    bf16_t* __restrict__ Qb, bf16_t* __restrict__ Kb, bf16_t* __restrict__ Vt) {
  __shared__ bf16_t As[BM][BK];
  __shared__ bf16_t Bs[BN][BK];

  const int tid = threadIdx.x;
  const int lane = tid & 31;
  const int wave = tid >> 5;
  const int wm = wave >> 1;             // 0..3
  const int wn = wave & 1;              // 0..1
  const int half = lane >> 4, lc = lane & 15;

  const int m0 = blockIdx.x * BM;
  const int which = blockIdx.y >> 3;    // 0=Q 1=K 2=V
  const int n0 = (blockIdx.y & 7) * BN;
  const float* W = (which == 0) ? Wq : (which == 1) ? Wk : Wv;

  v8f acc[2][4];
  #pragma unroll
  for (int i = 0; i < 2; ++i)
    #pragma unroll
    for (int j = 0; j < 4; ++j) acc[i][j] = (v8f)0.0f;

  for (int kt = 0; kt < E_DIM / BK; ++kt) {
    const int k0 = kt * BK;
    // stage A (x) and B (W) tiles, f32 -> bf16; 4 float4 per thread each.
    #pragma unroll
    for (int i = 0; i < 4; ++i) {
      int idx4 = tid * 4 + i;           // 0..1023
      int row = idx4 >> 3, c4 = idx4 & 7;
      float4 v = ((const float4*)(x + (size_t)(m0 + row) * E_DIM + k0))[c4];
      bf16_t* d = &As[row][c4 * 4];
      d[0] = f2bf(v.x); d[1] = f2bf(v.y); d[2] = f2bf(v.z); d[3] = f2bf(v.w);
    }
    #pragma unroll
    for (int i = 0; i < 4; ++i) {
      int idx4 = tid * 4 + i;
      int row = idx4 >> 3, c4 = idx4 & 7;
      float4 v = ((const float4*)(W + (size_t)(n0 + row) * E_DIM + k0))[c4];
      bf16_t* d = &Bs[row][c4 * 4];
      d[0] = f2bf(v.x); d[1] = f2bf(v.y); d[2] = f2bf(v.z); d[3] = f2bf(v.w);
    }
    if (kt + 1 < E_DIM / BK) {          // hint next tiles toward L2
      __builtin_prefetch(x + (size_t)(m0 + (tid >> 1)) * E_DIM + k0 + BK, 0, 1);
      __builtin_prefetch(W + (size_t)(n0 + (tid >> 1)) * E_DIM + k0 + BK, 0, 1);
    }
    __syncthreads();

    v16bf a[2], b[4];
    #pragma unroll
    for (int i = 0; i < 2; ++i) a[i] = frag_a(&As[wm * 32 + i * 16][0], BK, lane);
    #pragma unroll
    for (int j = 0; j < 4; ++j) b[j] = frag_b(&Bs[wn * 64 + j * 16][0], BK, lane);
    #pragma unroll
    for (int i = 0; i < 2; ++i)
      #pragma unroll
      for (int j = 0; j < 4; ++j) acc[i][j] = wmma_bf16(a[i], b[j], acc[i][j]);
    __syncthreads();
  }

  // scatter epilogue into per-head layouts (bf16)
  #pragma unroll
  for (int i = 0; i < 2; ++i) {
    #pragma unroll
    for (int j = 0; j < 4; ++j) {
      #pragma unroll
      for (int r = 0; r < 8; ++r) {
        int gm = m0 + wm * 32 + i * 16 + half * 8 + r;   // row in B*S
        int gn = n0 + wn * 64 + j * 16 + lc;             // col in E
        int bb = gm / SEQ, s = gm % SEQ;
        int h = gn / HDIM, d = gn % HDIM;
        bf16_t val = f2bf(acc[i][j][r]);
        size_t bh = (size_t)bb * NHEADS + h;
        if (which == 2)      Vt[(bh * HDIM + d) * SEQ + s] = val;
        else if (which == 1) Kb[(bh * SEQ + s) * HDIM + d] = val;
        else                 Qb[(bh * SEQ + s) * HDIM + d] = val;
      }
    }
  }
}

// ---------------------------------------------------------------------------
// Kernel 2: flash attention. Block = one (b,h) x 128-query tile, 8 waves;
// each wave owns 16 query rows (softmax stats stay inside a half-wave).
// K/V tiles staged with CDNA5 async global->LDS copies (ASYNCcnt).
// ---------------------------------------------------------------------------
#define BQ  128
#define BKV 128

__global__ __launch_bounds__(256) void attn(
    const bf16_t* __restrict__ Qb, const bf16_t* __restrict__ Kb,
    const bf16_t* __restrict__ Vt, bf16_t* __restrict__ Mb) {
  __shared__ bf16_t Ks[BKV][HDIM];        // 16 KB  key rows x d
  __shared__ bf16_t Vs[HDIM][BKV];        // 16 KB  d rows x key cols (V^T)
  __shared__ bf16_t Ps[8][16][BKV];       // 32 KB  per-wave P transpose staging

  const int tid = threadIdx.x;
  const int lane = tid & 31;
  const int wave = tid >> 5;
  const int half = lane >> 4, lc = lane & 15;
  const int q0 = blockIdx.x * BQ;
  const int bh = blockIdx.y;              // b*H + h
  const size_t qk_base = (size_t)bh * SEQ * HDIM;
  const size_t vt_base = (size_t)bh * HDIM * SEQ;
  const float scale = 0.125f;             // 1/sqrt(64)

  // This wave's Q fragments (16 rows x 64 d) straight from global (row-major).
  v16bf qf[2];
  {
    const bf16_t* qp = Qb + qk_base + (size_t)(q0 + wave * 16 + lc) * HDIM;
    int kb = half << 3;
    union { v16bf v; v8bf h[2]; } u0, u1;
    u0.h[0] = *(const v8bf*)(qp + kb);       u0.h[1] = *(const v8bf*)(qp + kb + 16);
    u1.h[0] = *(const v8bf*)(qp + 32 + kb);  u1.h[1] = *(const v8bf*)(qp + 48 + kb);
    qf[0] = u0.v; qf[1] = u1.v;
  }

  float m_run[8], l_run[8];
  v8f o[4];
  #pragma unroll
  for (int r = 0; r < 8; ++r) { m_run[r] = -1e30f; l_run[r] = 0.0f; }
  #pragma unroll
  for (int d = 0; d < 4; ++d) o[d] = (v8f)0.0f;

  for (int kv0 = 0; kv0 < SEQ; kv0 += BKV) {
    // stage K tile (contiguous 16KB) and V^T tile (64 rows x 256B) via
    // async global->LDS copies; each wave waits on its own ASYNCcnt, then
    // the workgroup barrier makes the whole tile visible.
    {
      const bf16_t* src = Kb + qk_base + (size_t)kv0 * HDIM;
      bf16_t* dst = &Ks[0][0];
      #pragma unroll
      for (int i = 0; i < 4; ++i) {
        int idx = tid + 256 * i;
        async_copy16(dst + idx * 8, src + idx * 8);
      }
    }
    #pragma unroll
    for (int i = 0; i < 4; ++i) {
      int idx = tid + 256 * i;
      int row = idx >> 4, c16 = idx & 15;
      async_copy16(&Vs[row][c16 * 8],
                   Vt + vt_base + (size_t)row * SEQ + kv0 + c16 * 8);
    }
    async_wait();
    __syncthreads();

    // S = Q K^T  (8 column tiles of 16, K-dim = 64 in two chunks)
    v8f sc[8];
    #pragma unroll
    for (int n = 0; n < 8; ++n) sc[n] = (v8f)0.0f;
    #pragma unroll
    for (int kk = 0; kk < 2; ++kk)
      #pragma unroll
      for (int n = 0; n < 8; ++n)
        sc[n] = wmma_bf16(qf[kk], frag_b(&Ks[n * 16][kk * 32], HDIM, lane), sc[n]);

    // online softmax (rows live in a 16-lane half-wave)
    float mnew[8], corr[8], rsum[8];
    #pragma unroll
    for (int r = 0; r < 8; ++r) {
      float mx = -1e30f;
      #pragma unroll
      for (int n = 0; n < 8; ++n) mx = fmaxf(mx, sc[n][r] * scale);
      #pragma unroll
      for (int off = 1; off < 16; off <<= 1) mx = fmaxf(mx, __shfl_xor(mx, off));
      mnew[r] = fmaxf(m_run[r], mx);
      corr[r] = __expf(m_run[r] - mnew[r]);
      rsum[r] = 0.0f;
    }
    #pragma unroll
    for (int d = 0; d < 4; ++d)
      #pragma unroll
      for (int r = 0; r < 8; ++r) o[d][r] *= corr[r];

    #pragma unroll
    for (int n = 0; n < 8; ++n)
      #pragma unroll
      for (int r = 0; r < 8; ++r) {
        float p = __expf(sc[n][r] * scale - mnew[r]);
        rsum[r] += p;
        Ps[wave][half * 8 + r][n * 16 + lc] = f2bf(p);   // transpose via LDS
      }
    #pragma unroll
    for (int r = 0; r < 8; ++r) {
      float s = rsum[r];
      #pragma unroll
      for (int off = 1; off < 16; off <<= 1) s += __shfl_xor(s, off);
      l_run[r] = l_run[r] * corr[r] + s;
      m_run[r] = mnew[r];
    }

    // O += P @ V   (per-wave LDS ops are in-order; no cross-wave use of Ps)
    #pragma unroll
    for (int kc = 0; kc < 4; ++kc) {
      v16bf pa = frag_a(&Ps[wave][0][kc * 32], BKV, lane);
      #pragma unroll
      for (int d = 0; d < 4; ++d)
        o[d] = wmma_bf16(pa, frag_b(&Vs[d * 16][kc * 32], BKV, lane), o[d]);
    }
    __syncthreads();
  }

  // normalize + write attention output as bf16 [B,S,E]
  const int bb = bh / NHEADS, h = bh % NHEADS;
  #pragma unroll
  for (int d = 0; d < 4; ++d)
    #pragma unroll
    for (int r = 0; r < 8; ++r) {
      int row = q0 + wave * 16 + half * 8 + r;
      Mb[((size_t)bb * SEQ + row) * E_DIM + h * HDIM + d * 16 + lc] =
          f2bf(o[d][r] / l_run[r]);
    }
}

// ---------------------------------------------------------------------------
// Kernel 3: conv1d(k=3,pad=1) as GEMM over K = 3 shifts x 1024 channels.
// out[b,s,eo] = bc[eo] + sum_k sum_ei M[b,s+k-1,ei] * WcT[k][eo][ei]
// bf16 tiles staged with async global->LDS; zero padding rows via ds_store.
// ---------------------------------------------------------------------------
__global__ __launch_bounds__(256) void conv_gemm(
    const bf16_t* __restrict__ Mb, const bf16_t* __restrict__ WcT,
    const float* __restrict__ bc, float* __restrict__ out) {
  __shared__ bf16_t As[BM][BK];
  __shared__ bf16_t Bs[BN][BK];

  const int tid = threadIdx.x;
  const int lane = tid & 31;
  const int wave = tid >> 5;
  const int wm = wave >> 1, wn = wave & 1;
  const int half = lane >> 4, lc = lane & 15;

  const int m0 = blockIdx.x * BM;       // rows over B*S (tile never straddles b)
  const int n0 = blockIdx.y * BN;
  const int bb = m0 / SEQ;
  const int sbase = m0 % SEQ;

  v8f acc[2][4];
  #pragma unroll
  for (int i = 0; i < 2; ++i)
    #pragma unroll
    for (int j = 0; j < 4; ++j) acc[i][j] = (v8f)0.0f;

  const int chunks = E_DIM / BK;        // 32 per shift
  for (int kt = 0; kt < 3 * chunks; ++kt) {
    const int shift = kt / chunks;      // 0,1,2
    const int k0 = (kt % chunks) * BK;
    // A tile: shifted rows of Mb (bf16 already), zero-padded at batch edges.
    #pragma unroll
    for (int i = 0; i < 2; ++i) {
      int idx = tid * 2 + i;            // 0..511 16B chunks
      int row = idx >> 2, c = idx & 3;
      int s = sbase + row + shift - 1;
      bf16_t* dst = &As[row][c * 8];
      if (s >= 0 && s < SEQ)
        async_copy16(dst, Mb + ((size_t)bb * SEQ + s) * E_DIM + k0 + c * 8);
      else {
        uint4 z = {0u, 0u, 0u, 0u};
        *(uint4*)dst = z;
      }
    }
    // B tile: WcT[shift] rows n0..n0+127 (contiguous bf16)
    #pragma unroll
    for (int i = 0; i < 2; ++i) {
      int idx = tid * 2 + i;
      int row = idx >> 2, c = idx & 3;
      async_copy16(&Bs[row][c * 8],
                   WcT + (size_t)shift * E_DIM * E_DIM +
                       (size_t)(n0 + row) * E_DIM + k0 + c * 8);
    }
    async_wait();
    __syncthreads();

    v16bf a[2], b[4];
    #pragma unroll
    for (int i = 0; i < 2; ++i) a[i] = frag_a(&As[wm * 32 + i * 16][0], BK, lane);
    #pragma unroll
    for (int j = 0; j < 4; ++j) b[j] = frag_b(&Bs[wn * 64 + j * 16][0], BK, lane);
    #pragma unroll
    for (int i = 0; i < 2; ++i)
      #pragma unroll
      for (int j = 0; j < 4; ++j) acc[i][j] = wmma_bf16(a[i], b[j], acc[i][j]);
    __syncthreads();
  }

  #pragma unroll
  for (int i = 0; i < 2; ++i)
    #pragma unroll
    for (int j = 0; j < 4; ++j)
      #pragma unroll
      for (int r = 0; r < 8; ++r) {
        int gm = m0 + wm * 32 + i * 16 + half * 8 + r;
        int gn = n0 + wn * 64 + j * 16 + lc;
        out[(size_t)gm * E_DIM + gn] = acc[i][j][r] + bc[gn];
      }
}

// ---------------------------------------------------------------------------
// launch
// ---------------------------------------------------------------------------
extern "C" void kernel_launch(void* const* d_in, const int* in_sizes, int n_in,
                              void* d_out, int out_size, void* d_ws, size_t ws_size,
                              hipStream_t stream) {
  const float* x  = (const float*)d_in[0];
  const float* Wq = (const float*)d_in[1];
  const float* Wk = (const float*)d_in[2];
  const float* Wv = (const float*)d_in[3];
  const float* Wc = (const float*)d_in[4];
  const float* bc = (const float*)d_in[5];
  float* out = (float*)d_out;

  // workspace carve-up (~70 MB of bf16)
  const size_t nQKV = (size_t)BATCH * NHEADS * SEQ * HDIM;   // 8,388,608
  const size_t nM   = (size_t)BATCH * SEQ * E_DIM;           // 8,388,608
  bf16_t* Qb  = (bf16_t*)d_ws;
  bf16_t* Kb  = Qb + nQKV;
  bf16_t* Vt  = Kb + nQKV;
  bf16_t* Mb  = Vt + nQKV;
  bf16_t* WcT = Mb + nM;                                      // 3*E*E bf16

  prep_wc<<<dim3((E_DIM * E_DIM) / 256), dim3(256), 0, stream>>>(Wc, WcT);
  qkv_gemm<<<dim3(MROWS / BM, 24), dim3(256), 0, stream>>>(x, Wq, Wk, Wv,
                                                           Qb, Kb, Vt);
  attn<<<dim3(SEQ / BQ, BATCH * NHEADS), dim3(256), 0, stream>>>(Qb, Kb, Vt, Mb);
  conv_gemm<<<dim3(MROWS / BM, E_DIM / BN), dim3(256), 0, stream>>>(Mb, WcT, bc,
                                                                    out);
}
